// RTEncoderOnly_36447092474056
// MI455X (gfx1250) — compile-verified, matrix operand order checked by reading, not run on hardware
//
#include <hip/hip_runtime.h>

#define BB 2
#define SS 1024
#define KFO 8
#define DD 256
#define HH 8
#define DHH 32
#define FFF 1024
#define DCC 384
#define DTT 384
#define NBB 2

typedef __attribute__((ext_vector_type(16))) __bf16 v16bf;
typedef __attribute__((ext_vector_type(8)))  float  v8f;

// K-index of element e (0..15) of a 16-elem bf16 fragment for lane-half lh,
// per CDNA5 ISA 16-bit A-matrix 16x32 layout (CDNA5 ISA 7.12.2).
__device__ __forceinline__ int kmap(int e, int lh) {
  int r = e >> 1, lohi = e & 1;
  int base = (r < 4) ? (2 * r) : (16 + 2 * (r - 4));
  return base + lh * 8 + lohi;
}

__device__ __forceinline__ v8f wmma_bf16(v16bf a, v16bf b, v8f c) {
  return __builtin_amdgcn_wmma_f32_16x16x32_bf16(false, a, false, b, (short)0, c,
                                                 false, false);
}

// LDS byte offset of a generic (flat) pointer to __shared__: the LDS aperture
// maps addr[31:0] to the wave's LDS offset (ISA 10.2), so truncation suffices.
__device__ __forceinline__ unsigned lds_lo(const void* p) {
  return (unsigned)(unsigned long long)p;
}

// Async global->LDS 16-byte copy (per lane), tracked by ASYNCcnt.
__device__ __forceinline__ void async_b128(unsigned lds, const float* g) {
  asm volatile("global_load_async_to_lds_b128 %0, %1, off"
               :: "v"(lds), "v"(g) : "memory");
}
#define WAIT_ASYNC(n) asm volatile("s_wait_asynccnt %0" :: "n"(n) : "memory")

// ---------------- mask construction: pack 4 masks into bits 0..3 --------------
__global__ void k_masks(const int* __restrict__ node, const int* __restrict__ f2p,
                        const int* __restrict__ coln, const int* __restrict__ tabn,
                        const unsigned char* __restrict__ pad,
                        unsigned char* __restrict__ out) {
  const int bq = blockIdx.x;              // b*S + q
  const int b = bq / SS;
  const int nq = node[bq], cq = coln[bq], tq = tabn[bq];
  const bool pq = (pad[bq] == 0);
  int fq[KFO];
#pragma unroll
  for (int j = 0; j < KFO; ++j) fq[j] = f2p[bq * KFO + j];
  for (int k = threadIdx.x; k < SS; k += blockDim.x) {
    const int idx = b * SS + k;
    const bool pp = pq && (pad[idx] == 0);
    const int nk = node[idx];
    bool kvin = false, qin = false;
#pragma unroll
    for (int j = 0; j < KFO; ++j) {
      kvin |= (nk == fq[j]);
      qin  |= (nq == f2p[idx * KFO + j]);
    }
    const bool same = (nq == nk);
    const bool sct = (cq == coln[idx]) && (tq == tabn[idx]);
    unsigned char m = 0;
    if ((same || kvin) && pp) m |= 1;  // feat
    if (qin && pp)            m |= 2;  // nbr
    if (sct && pp)            m |= 4;  // col
    if (pp)                   m |= 8;  // full
    out[(size_t)bq * SS + k] = m;
  }
}

// ---------------- block LayerNorm helper (blockDim == DD) --------------------
__device__ __forceinline__ float blk_ln(float y, float* red, int j) {
  red[j] = y; __syncthreads();
  for (int s = DD / 2; s > 0; s >>= 1) { if (j < s) red[j] += red[j + s]; __syncthreads(); }
  const float mean = red[0] * (1.f / DD); __syncthreads();
  const float d = y - mean;
  red[j] = d * d; __syncthreads();
  for (int s = DD / 2; s > 0; s >>= 1) { if (j < s) red[j] += red[j + s]; __syncthreads(); }
  const float var = red[0] * (1.f / DD); __syncthreads();
  return d * rsqrtf(var + 1e-5f);
}

// ---------------- cell-value encoding ----------------------------------------
__global__ void k_encode(const unsigned char* __restrict__ ispad,
                         const int* __restrict__ sem,
                         const unsigned char* __restrict__ mflag,
                         const float* __restrict__ cv, const float* __restrict__ nv,
                         const float* __restrict__ tv,
                         const float* __restrict__ Wc, const float* __restrict__ bc,
                         const float* __restrict__ Wn, const float* __restrict__ bn,
                         const float* __restrict__ Wt, const float* __restrict__ bt,
                         const float* __restrict__ gc, const float* __restrict__ bec,
                         const float* __restrict__ gn, const float* __restrict__ ben,
                         const float* __restrict__ gt, const float* __restrict__ bet,
                         const float* __restrict__ men, const float* __restrict__ met,
                         float* __restrict__ x) {
  __shared__ float sval[DCC];
  __shared__ float red[DD];
  const int tok = blockIdx.x, j = threadIdx.x;
  const bool notpad = (ispad[tok] == 0);
  const bool mk = (mflag[tok] != 0);
  const int st = sem[tok];

  for (int c = j; c < DCC; c += DD) sval[c] = cv[(size_t)tok * DCC + c];
  __syncthreads();
  float y = bc[j];
  for (int c = 0; c < DCC; ++c) y += sval[c] * Wc[c * DD + j];
  float xv = (blk_ln(y, red, j) * gc[j] + bec[j]) * (notpad ? 1.f : 0.f);

  const float yn = nv[tok] * Wn[j] + bn[j];
  const float lnn = blk_ln(yn, red, j) * gn[j] + ben[j];
  const bool seln = (st == 0) && notpad;
  if (seln && !mk) xv += lnn;
  if (seln &&  mk) xv += men[j];

  __syncthreads();
  for (int c = j; c < DTT; c += DD) sval[c] = tv[(size_t)tok * DTT + c];
  __syncthreads();
  float yt = bt[j];
  for (int c = 0; c < DTT; ++c) yt += sval[c] * Wt[c * DD + j];
  const float lnt = blk_ln(yt, red, j) * gt[j] + bet[j];
  const bool selt = (st == 1) && notpad;
  if (selt && !mk) xv += lnt;
  if (selt &&  mk) xv += met[j];

  x[(size_t)tok * DD + j] = xv;
}

// ---------------- standalone LayerNorm ---------------------------------------
__global__ void k_ln(const float* __restrict__ x, const float* __restrict__ g,
                     const float* __restrict__ b, float* __restrict__ o) {
  __shared__ float red[DD];
  const int tok = blockIdx.x, j = threadIdx.x;
  const float v = x[(size_t)tok * DD + j];
  o[(size_t)tok * DD + j] = blk_ln(v, red, j) * g[j] + b[j];
}

// ---------------- bf16 WMMA GEMM: C(M,N) = A(M,K)*W(K,N) [+bias] [+C] --------
// One wave per 64x16 output tile: 4 accumulators, W fragment reused 4x.
// A (64x32) and W (32x16) fp32 tiles staged via double-buffered async
// global->LDS B128 copies (coalesced, overlapped with WMMA).
template <bool BIAS, bool ACC>
__global__ void k_gemm(const float* __restrict__ A, const float* __restrict__ W,
                       const float* __restrict__ bias, float* __restrict__ C,
                       int N, int Kd) {
  __shared__ float ast[2][64 * 32];   // 16 KB
  __shared__ float wst[2][32 * 16];   //  4 KB
  const int nt = N >> 4;
  const int m0 = (blockIdx.x / nt) << 6;
  const int n0 = (blockIdx.x % nt) << 4;
  const int lane = threadIdx.x, lm = lane & 15, lh = lane >> 4;

  auto issue = [&](int k0, int buf) {
#pragma unroll
    for (int i = 0; i < 16; ++i) {                 // A: 64 rows x 32 f32
      const int f4 = i * 32 + lane;
      const int row = f4 >> 3, c4 = (f4 & 7) << 2;
      async_b128(lds_lo(&ast[buf][row * 32 + c4]),
                 &A[(size_t)(m0 + row) * Kd + k0 + c4]);
    }
#pragma unroll
    for (int i = 0; i < 4; ++i) {                  // W: 32 rows x 16 f32
      const int f4 = i * 32 + lane;
      const int row = f4 >> 2, c4 = (f4 & 3) << 2;
      async_b128(lds_lo(&wst[buf][row * 16 + c4]),
                 &W[(size_t)(k0 + row) * N + n0 + c4]);
    }
  };

  v8f acc[4] = {{}, {}, {}, {}};
  issue(0, 0);
  const int steps = Kd >> 5;
  for (int s = 0; s < steps; ++s) {
    const int buf = s & 1;
    if (s + 1 < steps) { issue((s + 1) << 5, buf ^ 1); WAIT_ASYNC(20); }
    else               { WAIT_ASYNC(0); }
    v16bf w;
#pragma unroll
    for (int e = 0; e < 16; ++e) w[e] = (__bf16)wst[buf][kmap(e, lh) * 16 + lm];
#pragma unroll
    for (int t = 0; t < 4; ++t) {
      v16bf a;
#pragma unroll
      for (int e = 0; e < 16; ++e)
        a[e] = (__bf16)ast[buf][(t * 16 + lm) * 32 + kmap(e, lh)];
      acc[t] = wmma_bf16(a, w, acc[t]);
    }
  }
#pragma unroll
  for (int t = 0; t < 4; ++t) {
#pragma unroll
    for (int r = 0; r < 8; ++r) {
      const int row = m0 + t * 16 + r + lh * 8, col = n0 + lm;
      float v = acc[t][r];
      if constexpr (BIAS) v += bias[col];
      if constexpr (ACC)  v += C[(size_t)row * N + col];
      C[(size_t)row * N + col] = v;
    }
  }
}

// ---------------- fused masked attention (one wave per b,h,16-row q-tile) ----
// K/V tiles staged through double-buffered async global->LDS copies.
__global__ void k_attn(const float* __restrict__ Q, const float* __restrict__ K,
                       const float* __restrict__ V,
                       const unsigned char* __restrict__ mb, int bit,
                       float* __restrict__ O) {
  extern __shared__ float smem[];
  float* sc  = smem;                        // 16*SS f32 score strip (64 KB)
  float* kst = smem + 16 * SS;              // 2 x (16 x 32) f32
  float* vst = kst + 2 * 16 * DHH;          // 2 x (32 x 32) f32
  const int qt = blockIdx.x % (SS / 16);
  const int h = (blockIdx.x / (SS / 16)) % HH;
  const int b = blockIdx.x / ((SS / 16) * HH);
  const int lane = threadIdx.x, lm = lane & 15, lh = lane >> 4;
  const int hq = h * DHH;
  const float scale = 0.17677669529663687f;          // 1/sqrt(32)

  auto issue_k = [&](int kt, int buf) {
#pragma unroll
    for (int i = 0; i < 4; ++i) {                   // 16 keys x 32 f32
      const int f4 = i * 32 + lane;
      const int row = f4 >> 3, c4 = (f4 & 7) << 2;
      async_b128(lds_lo(&kst[buf * 512 + row * 32 + c4]),
                 &K[((size_t)(b * SS + kt * 16 + row)) * DD + hq + c4]);
    }
  };
  auto issue_v = [&](int kc, int buf) {
#pragma unroll
    for (int i = 0; i < 8; ++i) {                   // 32 keys x 32 f32
      const int f4 = i * 32 + lane;
      const int row = f4 >> 3, c4 = (f4 & 7) << 2;
      async_b128(lds_lo(&vst[buf * 1024 + row * 32 + c4]),
                 &V[((size_t)(b * SS + kc * 32 + row)) * DD + hq + c4]);
    }
  };

  // Q fragment (16 rows x DH=32), resident for the whole strip
  v16bf qf;
  const int qrow_lane = qt * 16 + lm;
#pragma unroll
  for (int e = 0; e < 16; ++e)
    qf[e] = (__bf16)Q[((size_t)(b * SS + qrow_lane)) * DD + hq + kmap(e, lh)];

  // scores: 64 WMMAs, K tiles via pipelined async copies
  issue_k(0, 0);
  for (int kt = 0; kt < SS / 16; ++kt) {
    const int buf = kt & 1;
    if (kt + 1 < SS / 16) { issue_k(kt + 1, buf ^ 1); WAIT_ASYNC(4); }
    else                  { WAIT_ASYNC(0); }
    v16bf kf;
#pragma unroll
    for (int e = 0; e < 16; ++e)
      kf[e] = (__bf16)kst[buf * 512 + lm * 32 + kmap(e, lh)];
    v8f c = {};
    c = wmma_bf16(qf, kf, c);
#pragma unroll
    for (int r = 0; r < 8; ++r) {
      const int row = r + lh * 8;
      const int col = kt * 16 + lm;
      const int qrow = qt * 16 + row;
      const unsigned char m = mb[((size_t)(b * SS + qrow)) * SS + col];
      sc[row * SS + col] = ((m >> bit) & 1) ? c[r] * scale : -1e9f;
    }
  }
  __syncthreads();

  // softmax: lane handles row lm, column half lh; pair combined via shfl_xor(16)
  {
    const int c0 = lh * (SS / 2);
    float mx = -3.0e38f;
    for (int c = 0; c < SS / 2; ++c) mx = fmaxf(mx, sc[lm * SS + c0 + c]);
    mx = fmaxf(mx, __shfl_xor(mx, 16, 32));
    float sum = 0.f;
    for (int c = 0; c < SS / 2; ++c) {
      const float e = __expf(sc[lm * SS + c0 + c] - mx);
      sc[lm * SS + c0 + c] = e;
      sum += e;
    }
    sum += __shfl_xor(sum, 16, 32);
    const float inv = (mx > -1e8f) ? (1.f / sum) : 0.f;  // fully-masked row -> 0
    for (int c = 0; c < SS / 2; ++c) sc[lm * SS + c0 + c] *= inv;
  }
  __syncthreads();

  // P @ V : 32 key-chunks x 2 DH-halves = 64 WMMAs, V tiles pipelined
  v8f o0 = {}, o1 = {};
  issue_v(0, 0);
  for (int kc = 0; kc < SS / 32; ++kc) {
    const int buf = kc & 1;
    if (kc + 1 < SS / 32) { issue_v(kc + 1, buf ^ 1); WAIT_ASYNC(8); }
    else                  { WAIT_ASYNC(0); }
    v16bf pf, v0, v1;
#pragma unroll
    for (int e = 0; e < 16; ++e) {
      const int kk = kmap(e, lh);
      pf[e] = (__bf16)sc[lm * SS + kc * 32 + kk];
      v0[e] = (__bf16)vst[buf * 1024 + kk * 32 + lm];
      v1[e] = (__bf16)vst[buf * 1024 + kk * 32 + 16 + lm];
    }
    o0 = wmma_bf16(pf, v0, o0);
    o1 = wmma_bf16(pf, v1, o1);
  }
#pragma unroll
  for (int r = 0; r < 8; ++r) {
    const int qrow = qt * 16 + r + lh * 8;
    float* orow = &O[((size_t)(b * SS + qrow)) * DD + hq];
    orow[lm] = o0[r];
    orow[16 + lm] = o1[r];
  }
}

// ---------------- gelu (tanh approx, jax.nn.gelu default) --------------------
__global__ void k_gelu(float* __restrict__ p, int n) {
  const int i = blockIdx.x * blockDim.x + threadIdx.x;
  if (i < n) {
    const float v = p[i];
    const float t = 0.7978845608028654f * (v + 0.044715f * v * v * v);
    p[i] = 0.5f * v * (1.f + tanhf(t));
  }
}

extern "C" void kernel_launch(void* const* d_in, const int* in_sizes, int n_in,
                              void* d_out, int out_size, void* d_ws, size_t ws_size,
                              hipStream_t stream) {
  (void)in_sizes; (void)n_in; (void)out_size; (void)ws_size;
  const int*           node  = (const int*)d_in[0];
  const int*           f2p   = (const int*)d_in[1];
  const int*           coln  = (const int*)d_in[2];
  const int*           tabn  = (const int*)d_in[3];
  const unsigned char* ispad = (const unsigned char*)d_in[4];
  const int*           sem   = (const int*)d_in[5];
  const unsigned char* mflag = (const unsigned char*)d_in[6];
  const float* cv  = (const float*)d_in[7];
  const float* nv  = (const float*)d_in[8];
  const float* tv  = (const float*)d_in[9];
  const float* Wc  = (const float*)d_in[10];
  const float* bc  = (const float*)d_in[11];
  const float* Wn  = (const float*)d_in[12];
  const float* bn  = (const float*)d_in[13];
  const float* Wt  = (const float*)d_in[14];
  const float* bt  = (const float*)d_in[15];
  const float* gc  = (const float*)d_in[16];
  const float* bec = (const float*)d_in[17];
  const float* gn  = (const float*)d_in[18];
  const float* ben = (const float*)d_in[19];
  const float* gt  = (const float*)d_in[20];
  const float* bet = (const float*)d_in[21];
  const float* men = (const float*)d_in[22];
  const float* met = (const float*)d_in[23];
  const float* Wq  = (const float*)d_in[24];
  const float* Wk  = (const float*)d_in[25];
  const float* Wv  = (const float*)d_in[26];
  const float* Wo  = (const float*)d_in[27];
  const float* l1g = (const float*)d_in[28];
  const float* l1b = (const float*)d_in[29];
  const float* l2g = (const float*)d_in[30];
  const float* l2b = (const float*)d_in[31];
  const float* W1  = (const float*)d_in[32];
  const float* b1  = (const float*)d_in[33];
  const float* W2  = (const float*)d_in[34];
  const float* b2  = (const float*)d_in[35];
  const float* gout= (const float*)d_in[36];
  const float* bout= (const float*)d_in[37];

  const int T = BB * SS;                 // 2048 tokens
  float* x  = (float*)d_ws;
  float* h  = x  + (size_t)T * DD;
  float* qb = h  + (size_t)T * DD;
  float* kb = qb + (size_t)T * DD;
  float* vb = kb + (size_t)T * DD;
  float* ob = vb + (size_t)T * DD;
  float* ff = ob + (size_t)T * DD;
  unsigned char* mbts = (unsigned char*)(ff + (size_t)T * FFF);

  k_masks<<<T, 256, 0, stream>>>(node, f2p, coln, tabn, ispad, mbts);
  k_encode<<<T, DD, 0, stream>>>(ispad, sem, mflag, cv, nv, tv, Wc, bc, Wn, bn,
                                 Wt, bt, gc, bec, gn, ben, gt, bet, men, met, x);

  const int gProj = (T / 64) * (DD / 16);     // 64x16 tiles
  const int gFfn1 = (T / 64) * (FFF / 16);
  const size_t attn_lds = (16 * SS + 2 * 16 * DHH + 2 * 32 * DHH) * sizeof(float);
  for (int i = 0; i < NBB; ++i) {
    k_ln<<<T, DD, 0, stream>>>(x, l1g + i * DD, l1b + i * DD, h);
    for (int l = 0; l < 4; ++l) {
      const size_t wOff = (size_t)(i * 4 + l) * DD * DD;
      k_gemm<false, false><<<gProj, 32, 0, stream>>>(h, Wq + wOff, nullptr, qb, DD, DD);
      k_gemm<false, false><<<gProj, 32, 0, stream>>>(h, Wk + wOff, nullptr, kb, DD, DD);
      k_gemm<false, false><<<gProj, 32, 0, stream>>>(h, Wv + wOff, nullptr, vb, DD, DD);
      k_attn<<<BB * HH * (SS / 16), 32, attn_lds, stream>>>(qb, kb, vb, mbts, l, ob);
      k_gemm<false, true><<<gProj, 32, 0, stream>>>(ob, Wo + wOff, nullptr, x, DD, DD);
    }
    k_ln<<<T, DD, 0, stream>>>(x, l2g + i * DD, l2b + i * DD, h);
    k_gemm<true, false><<<gFfn1, 32, 0, stream>>>(h, W1 + (size_t)i * DD * FFF,
                                                  b1 + i * FFF, ff, FFF, DD);
    const int ne = T * FFF;
    k_gelu<<<(ne + 255) / 256, 256, 0, stream>>>(ff, ne);
    k_gemm<true, true><<<gProj, 32, 0, stream>>>(ff, W2 + (size_t)i * FFF * DD,
                                                 b2 + i * DD, x, DD, FFF);
  }
  k_ln<<<T, DD, 0, stream>>>(x, gout, bout, (float*)d_out);
}
